// LinearAttention_17334488007434
// MI455X (gfx1250) — compile-verified, matrix operand order checked by reading, not run on hardware
//
#include <hip/hip_runtime.h>

// ---------------------------------------------------------------------------
// LinearAttention fused pipeline for gfx1250 (MI455X), wave32 + WMMA bf16.
// b=16, c=128, heads=4, dim_head=32, n=128*128=16384.
// ---------------------------------------------------------------------------

typedef __attribute__((ext_vector_type(16))) __bf16 v16bf;
typedef __attribute__((ext_vector_type(8)))  float  v8f;

struct Frag { union { v16bf v; uint4 q[2]; }; };

static __device__ inline Frag ld_frag(const unsigned short* p) {
  // 16 bf16 for this lane: bytes [0,16) and [32,48) relative to p (p is 16B aligned)
  Frag f;
  f.q[0] = *(const uint4*)(p);
  f.q[1] = *(const uint4*)(p + 16);
  return f;
}

static __device__ inline v8f wmma_bf16(const Frag& a, const Frag& b, v8f c) {
  return __builtin_amdgcn_wmma_f32_16x16x32_bf16(false, a.v, false, b.v,
                                                 (short)0, c, false, false);
}

static __device__ inline unsigned short f2bf(float f) {  // RNE float->bf16
  unsigned u = __float_as_uint(f);
  u += 0x7fffu + ((u >> 16) & 1u);
  return (unsigned short)(u >> 16);
}

// monotonic float<->uint encoding for atomicMax on signed floats
static __device__ inline unsigned encmax(float f) {
  unsigned u = __float_as_uint(f);
  return (u & 0x80000000u) ? ~u : (u | 0x80000000u);
}
static __device__ inline float decmax(unsigned u) {
  u = (u & 0x80000000u) ? (u & 0x7fffffffu) : ~u;
  return __uint_as_float(u);
}

#define NTOT   16384
#define XSTR   136   // ushorts per LDS row (272 B: 16B aligned, bank-spread)
#define YSTR   129   // floats per LDS row for f32 tiles

// ---------------------------------------------------------------------------
// init: w_qkv -> bf16, zero Cctx/lsum, kmax = enc(-inf)
// grid 256 x 256 threads (65536 lanes)
// ---------------------------------------------------------------------------
__global__ __launch_bounds__(256) void k_init(
    const float* __restrict__ w_qkv, unsigned short* __restrict__ wqkv_bf,
    float* __restrict__ Cctx, float* __restrict__ lsum, unsigned* __restrict__ kmaxU)
{
  int idx = blockIdx.x * 256 + threadIdx.x;           // 0..65535
  if (idx < 384 * 128) wqkv_bf[idx] = f2bf(w_qkv[idx]);
  Cctx[idx] = 0.0f;                                   // 16*4*32*32 = 65536
  if (idx < 16 * 128) { lsum[idx] = 0.0f; kmaxU[idx] = 0x007FFFFFu; }
}

// ---------------------------------------------------------------------------
// pass A: k = W_k @ x per 128-col tile (WMMA), global row-max -> kmaxU[b][128]
// grid (128, 16), 256 threads (8 waves)
// ---------------------------------------------------------------------------
__global__ __launch_bounds__(256) void k_kmax(
    const float* __restrict__ x, const unsigned short* __restrict__ wqkv,
    unsigned* __restrict__ kmaxU)
{
  __shared__ __align__(16) unsigned short xb[128 * XSTR];
  const int t = threadIdx.x, w = t >> 5, ln = t & 31;
  const int b = blockIdx.y, n0 = blockIdx.x * 128;
  const float* xg = x + ((size_t)b * 128) * NTOT + n0;

  { // stage x tile transposed to [n][c] bf16
    int n = t & 127, ch = t >> 7;
    for (int it = 0; it < 64; ++it) {
      int c = it * 2 + ch;
      if (it + 2 < 64) __builtin_prefetch(&xg[(size_t)(c + 4) * NTOT + n], 0, 1);
      xb[n * XSTR + c] = f2bf(xg[(size_t)c * NTOT + n]);
    }
  }
  __syncthreads();

  const int lr = ln & 15, hi = ln >> 4, sel = hi * 8, mrow = w * 16;
  v8f acc[8];
  for (int j = 0; j < 8; ++j)
    for (int r = 0; r < 8; ++r) acc[j][r] = 0.0f;

#pragma unroll
  for (int ks = 0; ks < 4; ++ks) {
    Frag a = ld_frag(wqkv + (size_t)(128 + mrow + lr) * 128 + ks * 32 + sel); // k rows
#pragma unroll
    for (int j = 0; j < 8; ++j) {
      Frag bb = ld_frag(xb + (j * 16 + lr) * XSTR + ks * 32 + sel);
      acc[j] = wmma_bf16(a, bb, acc[j]);
    }
  }

#pragma unroll
  for (int r = 0; r < 8; ++r) {
    float m = acc[0][r];
#pragma unroll
    for (int j = 1; j < 8; ++j) m = fmaxf(m, acc[j][r]);
    m = fmaxf(m, __shfl_xor(m, 1, 32));
    m = fmaxf(m, __shfl_xor(m, 2, 32));
    m = fmaxf(m, __shfl_xor(m, 4, 32));
    m = fmaxf(m, __shfl_xor(m, 8, 32));
    if (lr == 0) {
      int row = mrow + r + hi * 8;                      // k channel 0..127
      atomicMax(&kmaxU[b * 128 + row], encmax(m));
    }
  }
}

// ---------------------------------------------------------------------------
// pass B: kv = W_kv @ x (WMMA), exp(k-m), v/N; accumulate lsum + per-head
// 32x32 context partials (scalar, tiny) with f32 atomics.
// grid (32, 16): each block walks 8 sub-tiles of 64 columns (512 cols total)
// ---------------------------------------------------------------------------
__global__ __launch_bounds__(256) void k_ctx(
    const float* __restrict__ x, const unsigned short* __restrict__ wqkv,
    const unsigned* __restrict__ kmaxU, float* __restrict__ lsum,
    float* __restrict__ Cctx)
{
  __shared__ __align__(16) unsigned short xb64[64 * XSTR];
  __shared__ float kvf[256 * 65];
  const int t = threadIdx.x, w = t >> 5, ln = t & 31;
  const int b = blockIdx.y, n0 = blockIdx.x * 512;
  const int lr = ln & 15, hi = ln >> 4, sel = hi * 8;
  const float* xg = x + ((size_t)b * 128) * NTOT;
  const float km = (t < 128) ? decmax(kmaxU[b * 128 + t]) : 0.0f;
  const float invN = 1.0f / (float)NTOT;

  float lacc = 0.0f;
  float cacc[16];
  for (int i = 0; i < 16; ++i) cacc[i] = 0.0f;

  for (int st = 0; st < 8; ++st) {
    const int nt = n0 + st * 64;
    { // stage 64-column x tile -> [n][c] bf16
      int n = t & 63, ch = t >> 6;
      for (int it = 0; it < 32; ++it) {
        int c = it * 4 + ch;
        xb64[n * XSTR + c] = f2bf(xg[(size_t)c * NTOT + nt + n]);
      }
    }
    __syncthreads();

    // kv[256 x 64] = W_kv[256 x 128] @ x_tile ; wave w owns rows 32w..32w+31
    v8f acc[8];
    for (int j = 0; j < 8; ++j)
      for (int r = 0; r < 8; ++r) acc[j][r] = 0.0f;
#pragma unroll
    for (int ks = 0; ks < 4; ++ks) {
      Frag a0 = ld_frag(wqkv + (size_t)(128 + w * 32 + lr) * 128 + ks * 32 + sel);
      Frag a1 = ld_frag(wqkv + (size_t)(128 + w * 32 + 16 + lr) * 128 + ks * 32 + sel);
#pragma unroll
      for (int j = 0; j < 4; ++j) {
        Frag bb = ld_frag(xb64 + (j * 16 + lr) * XSTR + ks * 32 + sel);
        acc[j]     = wmma_bf16(a0, bb, acc[j]);
        acc[4 + j] = wmma_bf16(a1, bb, acc[4 + j]);
      }
    }
#pragma unroll
    for (int rt = 0; rt < 2; ++rt)
#pragma unroll
      for (int j = 0; j < 4; ++j)
#pragma unroll
        for (int r = 0; r < 8; ++r)
          kvf[(w * 32 + rt * 16 + r + hi * 8) * 65 + j * 16 + lr] = acc[rt * 4 + j][r];
    __syncthreads();

    // rows 0..127: exp(k - max), accumulate row sums; rows 128..255: v/N
    {
      float* rr = &kvf[t * 65];
      if (t < 128) {
        for (int n = 0; n < 64; ++n) { float e = __expf(rr[n] - km); rr[n] = e; lacc += e; }
      } else {
        for (int n = 0; n < 64; ++n) rr[n] *= invN;
      }
    }
    __syncthreads();

    // context partials: C[h][d][e] += sum_n ek[32h+d][n] * vn[32h+e][n]
#pragma unroll
    for (int i = 0; i < 16; ++i) {
      int p = i * 256 + t;                // 0..4095
      int h = p >> 10, d = (p >> 5) & 31, e = p & 31;
      const float* ek = &kvf[(h * 32 + d) * 65];
      const float* vv = &kvf[(128 + h * 32 + e) * 65];
      float s = cacc[i];
      for (int n = 0; n < 64; ++n) s += ek[n] * vv[n];
      cacc[i] = s;
    }
    __syncthreads();
  }

  if (t < 128) atomicAdd(&lsum[b * 128 + t], lacc);
  for (int i = 0; i < 16; ++i) {
    int p = i * 256 + t;
    int h = p >> 10, d = (p >> 5) & 31, e = p & 31;
    atomicAdd(&Cctx[b * 4096 + h * 1024 + d * 32 + e], cacc[i]);
  }
}

// ---------------------------------------------------------------------------
// fold context + w_out + scale into per-batch F[128][128] (bf16):
//   F[o][32h+d] = scale * sum_e w_out[o][32h+e] * (Cctx[h][d][e] / lsum[32h+d])
// grid 16 x 256
// ---------------------------------------------------------------------------
__global__ __launch_bounds__(256) void k_fmat(
    const float* __restrict__ Cctx, const float* __restrict__ lsum,
    const float* __restrict__ w_out, unsigned short* __restrict__ Fb)
{
  __shared__ float ctx[4096];
  const int b = blockIdx.x, t = threadIdx.x;
  for (int i = 0; i < 16; ++i) {
    int idx = i * 256 + t;
    int h = idx >> 10, d = (idx >> 5) & 31;
    ctx[idx] = Cctx[b * 4096 + idx] / lsum[b * 128 + h * 32 + d];
  }
  __syncthreads();
  const float scale = 0.17677669529663687f;             // 1/sqrt(32)
  for (int i = 0; i < 64; ++i) {
    int fi = i * 256 + t;                               // 0..16383
    int o = fi >> 7, hd = fi & 127, h = hd >> 5, d = hd & 31;
    float s = 0.0f;
#pragma unroll
    for (int e = 0; e < 32; ++e)
      s += w_out[o * 128 + h * 32 + e] * ctx[h * 1024 + d * 32 + e];
    Fb[(size_t)b * 16384 + o * 128 + hd] = f2bf(s * scale);
  }
}

// ---------------------------------------------------------------------------
// main: q = W_q@x (WMMA) -> per-head softmax over d -> y = F@qsoft (WMMA)
//       -> +bias -> LayerNorm over channels -> store. grid (128, 16)
// ---------------------------------------------------------------------------
__global__ __launch_bounds__(256) void k_main(
    const float* __restrict__ x, const unsigned short* __restrict__ wqkv,
    const unsigned short* __restrict__ Fb, const float* __restrict__ b_out,
    const float* __restrict__ g, float* __restrict__ out)
{
  __shared__ __align__(16) unsigned short xb[128 * XSTR];
  __shared__ float yf[128 * YSTR];
  const int t = threadIdx.x, w = t >> 5, ln = t & 31;
  const int b = blockIdx.y, n0 = blockIdx.x * 128;
  const float* xg = x + ((size_t)b * 128) * NTOT + n0;

  { // stage x tile transposed to [n][c] bf16
    int n = t & 127, ch = t >> 7;
    for (int it = 0; it < 64; ++it) {
      int c = it * 2 + ch;
      if (it + 2 < 64) __builtin_prefetch(&xg[(size_t)(c + 4) * NTOT + n], 0, 1);
      xb[n * XSTR + c] = f2bf(xg[(size_t)c * NTOT + n]);
    }
  }
  __syncthreads();

  const int lr = ln & 15, hi = ln >> 4, sel = hi * 8, mrow = w * 16;

  // GEMM 1: q[128 x 128] = W_q @ x_tile
  v8f acc[8];
  for (int j = 0; j < 8; ++j)
    for (int r = 0; r < 8; ++r) acc[j][r] = 0.0f;
#pragma unroll
  for (int ks = 0; ks < 4; ++ks) {
    Frag a = ld_frag(wqkv + (size_t)(mrow + lr) * 128 + ks * 32 + sel);  // q rows
#pragma unroll
    for (int j = 0; j < 8; ++j) {
      Frag bb = ld_frag(xb + (j * 16 + lr) * XSTR + ks * 32 + sel);
      acc[j] = wmma_bf16(a, bb, acc[j]);
    }
  }
#pragma unroll
  for (int j = 0; j < 8; ++j)
#pragma unroll
    for (int r = 0; r < 8; ++r)
      yf[(mrow + r + hi * 8) * YSTR + j * 16 + lr] = acc[j][r];
  __syncthreads();

  // per-column per-head softmax over d (32), write bf16 qsoft to xb[n][hid]
  {
    int col = t & 127, h0 = (t >> 7) * 2;
    for (int hh = 0; hh < 2; ++hh) {
      int h = h0 + hh;
      const float* yc = &yf[(h * 32) * YSTR + col];
      float m = yc[0];
#pragma unroll
      for (int d = 1; d < 32; ++d) m = fmaxf(m, yc[d * YSTR]);
      float s = 0.0f;
#pragma unroll
      for (int d = 0; d < 32; ++d) s += __expf(yc[d * YSTR] - m);
      float inv = 1.0f / s;
#pragma unroll
      for (int d = 0; d < 32; ++d)
        xb[col * XSTR + h * 32 + d] = f2bf(__expf(yc[d * YSTR] - m) * inv);
    }
  }
  __syncthreads();

  // GEMM 2: y = F_b @ qsoft  (scale folded into F), + bias
  const unsigned short* Fbb = Fb + (size_t)b * 16384;
  for (int j = 0; j < 8; ++j)
    for (int r = 0; r < 8; ++r) acc[j][r] = 0.0f;
#pragma unroll
  for (int ks = 0; ks < 4; ++ks) {
    Frag a = ld_frag(Fbb + (size_t)(mrow + lr) * 128 + ks * 32 + sel);
#pragma unroll
    for (int j = 0; j < 8; ++j) {
      Frag bb = ld_frag(xb + (j * 16 + lr) * XSTR + ks * 32 + sel);
      acc[j] = wmma_bf16(a, bb, acc[j]);
    }
  }
  float br[8];
#pragma unroll
  for (int r = 0; r < 8; ++r) br[r] = b_out[mrow + r + hi * 8];
#pragma unroll
  for (int j = 0; j < 8; ++j)
#pragma unroll
    for (int r = 0; r < 8; ++r)
      yf[(mrow + r + hi * 8) * YSTR + j * 16 + lr] = acc[j][r] + br[r];
  __syncthreads();

  // LayerNorm over 128 channels per column, coalesced store
  if (t < 128) {
    int col = t;
    float s = 0.0f, s2 = 0.0f;
#pragma unroll 8
    for (int o = 0; o < 128; ++o) { float v = yf[o * YSTR + col]; s += v; s2 += v * v; }
    float mean = s * (1.0f / 128.0f);
    float var  = s2 * (1.0f / 128.0f) - mean * mean;
    float inv  = rsqrtf(var + 1e-5f);
    float* og = out + ((size_t)b * 128) * NTOT + n0 + col;
#pragma unroll 4
    for (int o = 0; o < 128; ++o)
      og[(size_t)o * NTOT] = (yf[o * YSTR + col] - mean) * inv * g[o];
  }
}

// ---------------------------------------------------------------------------
extern "C" void kernel_launch(void* const* d_in, const int* in_sizes, int n_in,
                              void* d_out, int out_size, void* d_ws, size_t ws_size,
                              hipStream_t stream) {
  (void)in_sizes; (void)n_in; (void)out_size; (void)ws_size;
  const float* x     = (const float*)d_in[0];
  const float* w_qkv = (const float*)d_in[1];
  const float* w_out = (const float*)d_in[2];
  const float* b_out = (const float*)d_in[3];
  const float* g     = (const float*)d_in[4];
  float* out = (float*)d_out;

  char* ws = (char*)d_ws;                       // < 1 MB total, all 16B aligned
  unsigned short* wqkv_bf = (unsigned short*)(ws);           //  98304 B
  unsigned short* Fb      = (unsigned short*)(ws + 98304);   // 524288 B
  unsigned*       kmaxU   = (unsigned*)      (ws + 622592);  //   8192 B
  float*          lsum    = (float*)         (ws + 630784);  //   8192 B
  float*          Cctx    = (float*)         (ws + 638976);  // 262144 B

  k_init<<<256, 256, 0, stream>>>(w_qkv, wqkv_bf, Cctx, lsum, kmaxU);
  k_kmax<<<dim3(128, 16), 256, 0, stream>>>(x, wqkv_bf, kmaxU);
  k_ctx <<<dim3(32, 16),  256, 0, stream>>>(x, wqkv_bf, kmaxU, lsum, Cctx);
  k_fmat<<<16, 256, 0, stream>>>(Cctx, lsum, w_out, Fb);
  k_main<<<dim3(128, 16), 256, 0, stream>>>(x, wqkv_bf, Fb, b_out, g, out);
}
